// NequipLayer_35244501631524
// MI455X (gfx1250) — compile-verified, matrix operand order checked by reading, not run on hardware
//
#include <hip/hip_runtime.h>
#include <hip/hip_bf16.h>

// ---------------------------------------------------------------------------
// NequIP-style layer for MI455X (gfx1250, wave32).
//  - edge radial-MLP runs on v_wmma_f32_16x16x32_f16 (16-edge tiles)
//  - WMMA fragments built branchlessly from zero-padded LDS tiles with
//    16B ds_load_b128 reads (K-map contiguous in groups of 8)
//  - radial tile + W1 staged via GLOBAL_LOAD_ASYNC_TO_LDS_B128 (ASYNCcnt path)
//    when the toolchain exposes the builtins; sync fallback otherwise
//  - gather/scatter phase uses L2-resident f32 atomics (tables fit in 192MB L2)
// ---------------------------------------------------------------------------

typedef __attribute__((ext_vector_type(16))) _Float16 v16h;
typedef __attribute__((ext_vector_type(8)))  _Float16 v8h;
typedef __attribute__((ext_vector_type(8)))  float    v8f;
typedef __attribute__((ext_vector_type(4)))  int      v4i;

#define MULC        16
#define RADIAL_INC  8
#define RADIAL_HIDC 64
#define NPATHS      5
#define WOUT        (NPATHS * MULC)   // 80
#define INV_SQRT_DEG 0.25f            // 1/sqrt(16)

#if __has_builtin(__builtin_amdgcn_global_load_async_to_lds_b128) && \
    __has_builtin(__builtin_amdgcn_s_wait_asynccnt)
#define USE_ASYNC_LDS 1
#else
#define USE_ASYNC_LDS 0
#endif

__device__ __forceinline__ float silu_f(float x) {
    return x * __builtin_amdgcn_rcpf(1.0f + __expf(-x));
}
__device__ __forceinline__ float sigmoid_f(float x) {
    return __builtin_amdgcn_rcpf(1.0f + __expf(-x));
}
__device__ __forceinline__ v16h cat8(v8h lo, v8h hi) {
    return __builtin_shufflevector(lo, hi, 0,1,2,3,4,5,6,7,8,9,10,11,12,13,14,15);
}

// ---------------------------------------------------------------------------
// Kernel 1: node input projections  s = ns @ W_in_s ;  v = einsum(nuc,uw->nwc)
// Also zeroes the per-node edge accumulators.
// ---------------------------------------------------------------------------
__global__ __launch_bounds__(256) void node_in_kernel(
    const float* __restrict__ node_scalars,   // [N,16]
    const float* __restrict__ node_vectors,   // [N,48]
    const float* __restrict__ W_in_s,         // [16,16]
    const float* __restrict__ W_in_v,         // [16,16]
    float* __restrict__ s_proj,               // [N,16]
    float* __restrict__ v_proj,               // [N,16,3]
    float* __restrict__ a_s,                  // [N,32]
    float* __restrict__ a_v,                  // [N,48,3]
    int N)
{
    int n = blockIdx.x * blockDim.x + threadIdx.x;
    if (n >= N) return;

    float s[MULC];
    float v[MULC * 3];
#pragma unroll
    for (int w = 0; w < MULC; ++w) s[w] = 0.0f;
#pragma unroll
    for (int i = 0; i < MULC * 3; ++i) v[i] = 0.0f;

    const float* ns = node_scalars + (long long)n * MULC;
    const float* nv = node_vectors + (long long)n * MULC * 3;

#pragma unroll
    for (int u = 0; u < MULC; ++u) {
        float su = ns[u];
        float v0 = nv[u * 3 + 0], v1 = nv[u * 3 + 1], v2 = nv[u * 3 + 2];
#pragma unroll
        for (int w = 0; w < MULC; ++w) {
            s[w]         += su * W_in_s[u * MULC + w];
            float wv      = W_in_v[u * MULC + w];
            v[w * 3 + 0] += v0 * wv;
            v[w * 3 + 1] += v1 * wv;
            v[w * 3 + 2] += v2 * wv;
        }
    }

    float* sp = s_proj + (long long)n * MULC;
    float* vp = v_proj + (long long)n * MULC * 3;
#pragma unroll
    for (int w = 0; w < MULC; ++w) sp[w] = s[w];
#pragma unroll
    for (int i = 0; i < MULC * 3; ++i) vp[i] = v[i];

    float* as = a_s + (long long)n * (2 * MULC);
    float* av = a_v + (long long)n * (3 * MULC * 3);
#pragma unroll
    for (int i = 0; i < 2 * MULC; ++i) as[i] = 0.0f;
#pragma unroll
    for (int i = 0; i < 3 * MULC * 3; ++i) av[i] = 0.0f;
}

// ---------------------------------------------------------------------------
// Kernel 2: edge pipeline. 4 waves / block, 16 edges / wave (64 edges / block).
// GEMM1: h = silu(radial @ W1 + b1)   [16x8]@[8x64]   WMMA, K zero-padded to 32
// GEMM2: w = h @ W2                   [16x64]@[64x80] WMMA, 2 K-steps of 32
// Then tensor-product + atomic scatter-add into receiver accumulators.
//
// WMMA 16-bit operand K-map (ISA 7.12.2): element i of lane (hi,p) holds
//   K = 8*hi + i          for i in [0,8)
//   K = 8*hi + i + 8      for i in [8,16)
// i.e. two contiguous 8-element (16B) runs -> two ds_load_b128 per fragment.
// ---------------------------------------------------------------------------
__global__ __launch_bounds__(128) void edge_kernel(
    const float* __restrict__ radial,      // [E,8]
    const float* __restrict__ edge_sh,     // [E,4]
    const int*   __restrict__ senders,     // [E]
    const int*   __restrict__ receivers,   // [E]
    const float* __restrict__ mlp_w1,      // [8,64]
    const float* __restrict__ mlp_b1,      // [64]
    const float* __restrict__ mlp_w2,      // [64,80]
    const float* __restrict__ s_proj,      // [N,16]
    const float* __restrict__ v_proj,      // [N,16,3]
    float* __restrict__ a_s,               // [N,32]
    float* __restrict__ a_v)               // [N,48,3]
{
    // All tiles zero-padded / transposed so fragment reads are unconditional.
    __shared__ __align__(16) _Float16 sW1t[RADIAL_HIDC][32];      // [col][Kpad]  4 KB
    __shared__ __align__(16) _Float16 sW2t[WOUT][RADIAL_HIDC];    // [col][K]    10 KB
    __shared__ __align__(16) float    sB1[RADIAL_HIDC];
    __shared__ __align__(16) _Float16 sA[4][MULC][32];            // [wave][edge][Kpad] 4 KB
    __shared__ __align__(16) _Float16 sH[4][MULC][RADIAL_HIDC];   // 8 KB
    __shared__ __align__(16) float    sWw[4][MULC][WOUT];         // 20 KB

    const int tid = threadIdx.x;
    const long long blkBase = (long long)blockIdx.x * 64;

#if USE_ASYNC_LDS
    __shared__ __align__(16) float sRadRaw[64 * RADIAL_INC];          // 2 KB
    __shared__ __align__(16) float sW1Raw[RADIAL_INC * RADIAL_HIDC];  // 2 KB

    // Kick off async DMA of the radial tile (512 f32) and W1 (512 f32):
    // one GLOBAL_LOAD_ASYNC_TO_LDS_B128 per thread each. These fly on
    // ASYNCcnt while we synchronously stage W2 below.
    {
        float* gr = (float*)(radial + blkBase * RADIAL_INC) + tid * 4;
        float* lw = (float*)mlp_w1 + tid * 4;
        __builtin_amdgcn_global_load_async_to_lds_b128(
            (__attribute__((address_space(1))) v4i*)gr,
            (__attribute__((address_space(3))) v4i*)(sRadRaw + tid * 4), 0, 0);
        __builtin_amdgcn_global_load_async_to_lds_b128(
            (__attribute__((address_space(1))) v4i*)lw,
            (__attribute__((address_space(3))) v4i*)(sW1Raw + tid * 4), 0, 0);
    }
#endif

    // ---- stage W2 transposed + bias (overlaps the async loads) -------------
    for (int i = tid; i < WOUT * RADIAL_HIDC; i += 128) {
        int col = i >> 6, k = i & 63;
        sW2t[col][k] = (_Float16)mlp_w2[k * WOUT + col];
    }
    if (tid < RADIAL_HIDC) sB1[tid] = mlp_b1[tid];

#if USE_ASYNC_LDS
    __builtin_amdgcn_s_wait_asynccnt(0);
    __syncthreads();
    // convert raw f32 LDS -> zero-padded f16 tiles (branchless select)
    for (int i = tid; i < RADIAL_HIDC * 32; i += 128) {
        int col = i >> 5, k = i & 31;
        float val = sW1Raw[(k & 7) * RADIAL_HIDC + col];
        sW1t[col][k] = (k < RADIAL_INC) ? (_Float16)val : (_Float16)0.0f;
    }
    for (int i = tid; i < 64 * 32; i += 128) {
        int e = i >> 5, k = i & 31;
        float val = sRadRaw[e * RADIAL_INC + (k & 7)];
        sA[e >> 4][e & 15][k] = (k < RADIAL_INC) ? (_Float16)val : (_Float16)0.0f;
    }
#else
    // ---- sync fallback: stage W1 + A tile directly from global ------------
    for (int i = tid; i < RADIAL_HIDC * 32; i += 128) {
        int col = i >> 5, k = i & 31;
        float val = mlp_w1[(k & 7) * RADIAL_HIDC + col];
        sW1t[col][k] = (k < RADIAL_INC) ? (_Float16)val : (_Float16)0.0f;
    }
    for (int i = tid; i < 64 * 32; i += 128) {
        int e = i >> 5, k = i & 31;
        float val = radial[(blkBase + e) * RADIAL_INC + (k & 7)];
        sA[e >> 4][e & 15][k] = (k < RADIAL_INC) ? (_Float16)val : (_Float16)0.0f;
    }
#endif
    __syncthreads();

    const int wave = tid >> 5;            // wave32
    const int lane = tid & 31;
    const int p    = lane & 15;
    const int hi   = lane >> 4;

    // ---- GEMM1 ------------------------------------------------------------
    const _Float16* arow = &sA[wave][p][0];
    v16h afrag = cat8(*(const v8h*)(arow + 8 * hi),
                      *(const v8h*)(arow + 16 + 8 * hi));

#pragma unroll
    for (int c = 0; c < RADIAL_HIDC / 16; ++c) {     // 4 column tiles of hidden
        const _Float16* brow = &sW1t[c * 16 + p][0];
        v16h bfrag = cat8(*(const v8h*)(brow + 8 * hi),
                          *(const v8h*)(brow + 16 + 8 * hi));
        v8f acc = {};
        acc = __builtin_amdgcn_wmma_f32_16x16x32_f16(false, afrag, false, bfrag,
                                                     (short)0, acc, false, false);
        // D layout: VGPR r, lanes 0-15 -> M=r ; lanes 16-31 -> M=r+8 ; N=p
#pragma unroll
        for (int r = 0; r < 8; ++r) {
            int m = r + 8 * hi;
            int n = c * 16 + p;
            sH[wave][m][n] = (_Float16)silu_f(acc[r] + sB1[n]);
        }
    }
    __syncthreads();

    // ---- GEMM2: K=64 -> two 32-wide WMMA steps -----------------------------
    const _Float16* hrow = &sH[wave][p][0];
    v16h a2[2];
#pragma unroll
    for (int kc = 0; kc < 2; ++kc)
        a2[kc] = cat8(*(const v8h*)(hrow + kc * 32 + 8 * hi),
                      *(const v8h*)(hrow + kc * 32 + 16 + 8 * hi));

#pragma unroll
    for (int c = 0; c < NPATHS; ++c) {               // 5 output tiles = 5 paths
        const _Float16* brow = &sW2t[c * 16 + p][0];
        v8f acc = {};
#pragma unroll
        for (int kc = 0; kc < 2; ++kc) {
            v16h bfrag = cat8(*(const v8h*)(brow + kc * 32 + 8 * hi),
                              *(const v8h*)(brow + kc * 32 + 16 + 8 * hi));
            acc = __builtin_amdgcn_wmma_f32_16x16x32_f16(false, a2[kc], false, bfrag,
                                                         (short)0, acc, false, false);
        }
#pragma unroll
        for (int r = 0; r < 8; ++r)
            sWw[wave][r + 8 * hi][c * 16 + p] = acc[r];
    }
    __syncthreads();

    // ---- scatter phase: 128 threads over 64 edges, 2 threads/edge ----------
    const int eLocal = tid >> 1;
    const int uBase  = (tid & 1) * 8;
    const long long e = blkBase + eLocal;
    const int wv = eLocal >> 4;
    const int m  = eLocal & 15;

    const int   snd = senders[e];
    const int   rcv = receivers[e];
    const float Y0 = edge_sh[e * 4 + 0];
    const float Yx = edge_sh[e * 4 + 1];
    const float Yy = edge_sh[e * 4 + 2];
    const float Yz = edge_sh[e * 4 + 3];

    const float* xs_row = s_proj + (long long)snd * MULC;
    const float* xv_row = v_proj + (long long)snd * MULC * 3;
    __builtin_prefetch(xs_row, 0, 0);   // global_prefetch_b8 into cache
    __builtin_prefetch(xv_row, 0, 0);

    float* as_row = a_s + (long long)rcv * (2 * MULC);
    float* av_row = a_v + (long long)rcv * (3 * MULC * 3);

#pragma unroll
    for (int uu = 0; uu < 8; ++uu) {
        int u = uBase + uu;
        float w0 = sWw[wv][m][0 * 16 + u];
        float w1 = sWw[wv][m][1 * 16 + u];
        float w2 = sWw[wv][m][2 * 16 + u];
        float w3 = sWw[wv][m][3 * 16 + u];
        float w4 = sWw[wv][m][4 * 16 + u];

        float xs = xs_row[u];
        float x0 = xv_row[u * 3 + 0];
        float x1 = xv_row[u * 3 + 1];
        float x2 = xv_row[u * 3 + 2];

        // scalar messages
        unsafeAtomicAdd(&as_row[u],        w0 * xs * Y0);
        unsafeAtomicAdd(&as_row[MULC + u], w1 * (x0 * Yx + x1 * Yy + x2 * Yz));

        // vector messages: xs*Yv, xv*Y0, cross(xv, Yv)
        float c0 = x1 * Yz - x2 * Yy;
        float c1 = x2 * Yx - x0 * Yz;
        float c2 = x0 * Yy - x1 * Yx;

        unsafeAtomicAdd(&av_row[(u)      * 3 + 0], w2 * xs * Yx);
        unsafeAtomicAdd(&av_row[(u)      * 3 + 1], w2 * xs * Yy);
        unsafeAtomicAdd(&av_row[(u)      * 3 + 2], w2 * xs * Yz);
        unsafeAtomicAdd(&av_row[(16 + u) * 3 + 0], w3 * x0 * Y0);
        unsafeAtomicAdd(&av_row[(16 + u) * 3 + 1], w3 * x1 * Y0);
        unsafeAtomicAdd(&av_row[(16 + u) * 3 + 2], w3 * x2 * Y0);
        unsafeAtomicAdd(&av_row[(32 + u) * 3 + 0], w4 * c0);
        unsafeAtomicAdd(&av_row[(32 + u) * 3 + 1], w4 * c1);
        unsafeAtomicAdd(&av_row[(32 + u) * 3 + 2], w4 * c2);
    }
}

// ---------------------------------------------------------------------------
// Kernel 3: node output projections + self-connection + gates + residual
// ---------------------------------------------------------------------------
__global__ __launch_bounds__(256) void node_out_kernel(
    const float* __restrict__ node_scalars,  // [N,16]
    const float* __restrict__ node_vectors,  // [N,48]
    const int*   __restrict__ species,       // [N]
    const float* __restrict__ W_out_s,       // [32,32]
    const float* __restrict__ W_out_v,       // [48,16]
    const float* __restrict__ sc_s,          // [S,16,32]
    const float* __restrict__ sc_v,          // [S,16,16]
    const float* __restrict__ a_s,           // [N,32]
    const float* __restrict__ a_v,           // [N,48,3]
    float* __restrict__ out,                 // [N,64]
    int N)
{
    int n = blockIdx.x * blockDim.x + threadIdx.x;
    if (n >= N) return;

    float pre_s[2 * MULC];
    float pre_v[MULC * 3];
#pragma unroll
    for (int i = 0; i < 2 * MULC; ++i) pre_s[i] = 0.0f;
#pragma unroll
    for (int i = 0; i < MULC * 3; ++i) pre_v[i] = 0.0f;

    // pre_s = (a_s * inv_sqrt_deg) @ W_out_s
    const float* as = a_s + (long long)n * (2 * MULC);
#pragma unroll
    for (int u = 0; u < 2 * MULC; ++u) {
        float au = as[u] * INV_SQRT_DEG;
#pragma unroll
        for (int w = 0; w < 2 * MULC; ++w)
            pre_s[w] += au * W_out_s[u * (2 * MULC) + w];
    }

    // pre_v = einsum('uc,uw->wc', a_v * inv_sqrt_deg, W_out_v)
    const float* av = a_v + (long long)n * (3 * MULC * 3);
#pragma unroll
    for (int u = 0; u < 3 * MULC; ++u) {
        float a0 = av[u * 3 + 0] * INV_SQRT_DEG;
        float a1 = av[u * 3 + 1] * INV_SQRT_DEG;
        float a2 = av[u * 3 + 2] * INV_SQRT_DEG;
#pragma unroll
        for (int w = 0; w < MULC; ++w) {
            float wv = W_out_v[u * MULC + w];
            pre_v[w * 3 + 0] += a0 * wv;
            pre_v[w * 3 + 1] += a1 * wv;
            pre_v[w * 3 + 2] += a2 * wv;
        }
    }

    // self-connection (species-indexed)
    const int spec = species[n];
    const float* scs = sc_s + (long long)spec * MULC * (2 * MULC);
    const float* scv = sc_v + (long long)spec * MULC * MULC;
    const float* ns  = node_scalars + (long long)n * MULC;
    const float* nv  = node_vectors + (long long)n * MULC * 3;
#pragma unroll
    for (int u = 0; u < MULC; ++u) {
        float su = ns[u];
        float v0 = nv[u * 3 + 0], v1 = nv[u * 3 + 1], v2 = nv[u * 3 + 2];
#pragma unroll
        for (int w = 0; w < 2 * MULC; ++w)
            pre_s[w] += su * scs[u * (2 * MULC) + w];
#pragma unroll
        for (int w = 0; w < MULC; ++w) {
            float wv = scv[u * MULC + w];
            pre_v[w * 3 + 0] += v0 * wv;
            pre_v[w * 3 + 1] += v1 * wv;
            pre_v[w * 3 + 2] += v2 * wv;
        }
    }

    // gated nonlinearity + residual ; out = [out_s (16) | out_v (48)]
    float* orow = out + (long long)n * (MULC + MULC * 3);
#pragma unroll
    for (int w = 0; w < MULC; ++w)
        orow[w] = silu_f(pre_s[w]) + ns[w];
#pragma unroll
    for (int w = 0; w < MULC; ++w) {
        float g = sigmoid_f(pre_s[MULC + w]);
        orow[MULC + w * 3 + 0] = pre_v[w * 3 + 0] * g + nv[w * 3 + 0];
        orow[MULC + w * 3 + 1] = pre_v[w * 3 + 1] * g + nv[w * 3 + 1];
        orow[MULC + w * 3 + 2] = pre_v[w * 3 + 2] * g + nv[w * 3 + 2];
    }
}

// ---------------------------------------------------------------------------
// Host launcher
// ---------------------------------------------------------------------------
extern "C" void kernel_launch(void* const* d_in, const int* in_sizes, int n_in,
                              void* d_out, int out_size, void* d_ws, size_t ws_size,
                              hipStream_t stream) {
    const float* node_scalars = (const float*)d_in[0];
    const float* node_vectors = (const float*)d_in[1];
    const float* edge_sh      = (const float*)d_in[2];
    const float* radial_emb   = (const float*)d_in[3];
    const int*   senders      = (const int*)  d_in[4];
    const int*   receivers    = (const int*)  d_in[5];
    const int*   species      = (const int*)  d_in[6];
    const float* W_in_s       = (const float*)d_in[7];
    const float* W_in_v       = (const float*)d_in[8];
    const float* mlp_w1       = (const float*)d_in[9];
    const float* mlp_b1       = (const float*)d_in[10];
    const float* mlp_w2       = (const float*)d_in[11];
    const float* W_out_s      = (const float*)d_in[12];
    const float* W_out_v      = (const float*)d_in[13];
    const float* sc_s         = (const float*)d_in[14];
    const float* sc_v         = (const float*)d_in[15];

    const int       N = in_sizes[0] / MULC;     // 50000
    const long long E = in_sizes[4];            // 800000 (divisible by 64)

    // workspace layout (all f32)
    float* s_proj = (float*)d_ws;                       // [N,16]
    float* v_proj = s_proj + (long long)N * MULC;       // [N,48]
    float* a_s    = v_proj + (long long)N * MULC * 3;   // [N,32]
    float* a_v    = a_s    + (long long)N * 2 * MULC;   // [N,144]

    node_in_kernel<<<(N + 255) / 256, 256, 0, stream>>>(
        node_scalars, node_vectors, W_in_s, W_in_v,
        s_proj, v_proj, a_s, a_v, N);

    edge_kernel<<<(int)(E / 64), 128, 0, stream>>>(
        radial_emb, edge_sh, senders, receivers,
        mlp_w1, mlp_b1, mlp_w2, s_proj, v_proj, a_s, a_v);

    node_out_kernel<<<(N + 255) / 256, 256, 0, stream>>>(
        node_scalars, node_vectors, species,
        W_out_s, W_out_v, sc_s, sc_v, a_s, a_v,
        (float*)d_out, N);
}
